// Node_Model_23562190586357
// MI455X (gfx1250) — matrix operand order repaired and checked by reading, not compiled, not run on hardware
//
#include <hip/hip_runtime.h>

typedef __attribute__((ext_vector_type(16))) __bf16 v16bf;
typedef __attribute__((ext_vector_type(8)))  __bf16 v8bf;
typedef __attribute__((ext_vector_type(4)))  __bf16 v4bf;
typedef __attribute__((ext_vector_type(8)))  float  v8f;
typedef __attribute__((ext_vector_type(4)))  float  v4f;

#define FX  128
#define FE  64
#define HID 128
#define K1  192   // FX + FE
#define K2  256   // FX + HID

// ---------------- CDNA5 async copy (global -> LDS), tracked by ASYNCcnt ----------------

__device__ __forceinline__ void async_copy16(unsigned lds_byte_addr, const void* gaddr) {
  // LDS[lds_byte_addr .. +15] = MEM[gaddr .. +15]
  asm volatile("global_load_async_to_lds_b128 %0, %1, off"
               :: "v"(lds_byte_addr), "v"(gaddr) : "memory");
}
__device__ __forceinline__ void async_wait0() {
  asm volatile("s_wait_asynccnt 0x0" ::: "memory");
}
// Low 32 bits of a flat pointer to LDS == LDS byte address (aperture in high bits).
__device__ __forceinline__ unsigned lds_addr_of(const void* p) {
  return (unsigned)(unsigned long long)p;
}

// ---------------- WMMA fragment helpers (layouts per CDNA5 ISA 7.12.2) ----------------

// A fragment: 16x32 bf16 tile at column kb of an LDS row-major [16 x lda] matrix.
__device__ __forceinline__ v16bf ldsA_frag(const __bf16* base, int lda, int kb, int lane) {
  int m    = lane & 15;
  int koff = (lane >> 4) << 3;             // 0 or 8
  const __bf16* p = base + m * lda + kb + koff;
  v8bf lo = *(const v8bf*)(p);             // K = kb+koff .. +7
  v8bf hi = *(const v8bf*)(p + 16);        // K = kb+16+koff .. +7
  return __builtin_shufflevector(lo, hi, 0,1,2,3,4,5,6,7,8,9,10,11,12,13,14,15);
}

// B fragment: 32x16 bf16 from LDS weights stored TRANSPOSED as Wt[n*ldk + k].
__device__ __forceinline__ v16bf ldsB_frag(const __bf16* Wt, int ldk, int kb, int nt, int lane) {
  int n  = (lane & 15) + (nt << 4);
  int kk = kb + ((lane >> 4) << 3);
  const __bf16* p = Wt + n * ldk + kk;
  v8bf lo = *(const v8bf*)(p);
  v8bf hi = *(const v8bf*)(p + 16);
  return __builtin_shufflevector(lo, hi, 0,1,2,3,4,5,6,7,8,9,10,11,12,13,14,15);
}

__device__ __forceinline__ v8f wmma_bf16(v16bf a, v16bf b, v8f c) {
  return __builtin_amdgcn_wmma_f32_16x16x32_bf16(false, a, false, b, (short)0, c, false, false);
}

__device__ __forceinline__ v4bf cvt4(v4f v) {
  v4bf r = { (__bf16)v.x, (__bf16)v.y, (__bf16)v.z, (__bf16)v.w };
  return r;
}

// ---------------- Kernel 1: prep (bf16 copies of x + transposed weights; zero scatter) ----

__global__ void prep_kernel(const float* __restrict__ x,
                            const float* __restrict__ W1a, const float* __restrict__ W1b,
                            const float* __restrict__ W2a, const float* __restrict__ W2b,
                            __bf16* __restrict__ xb,
                            __bf16* __restrict__ W1aT, __bf16* __restrict__ W1bT,
                            __bf16* __restrict__ W2aT, __bf16* __restrict__ W2bT,
                            float* __restrict__ sums, float* __restrict__ cnt, int N)
{
  long tid    = (long)blockIdx.x * blockDim.x + threadIdx.x;
  long stride = (long)gridDim.x * blockDim.x;
  long nx = (long)N * FX;                  // == N*HID since FX==HID
  for (long i = tid; i < nx; i += stride) { xb[i] = (__bf16)x[i]; sums[i] = 0.0f; }
  for (long i = tid; i < N; i += stride) cnt[i] = 0.0f;
  for (long i = tid; i < K1 * HID; i += stride) {
    int k = (int)(i >> 7), n = (int)(i & 127);
    W1aT[n * K1 + k] = (__bf16)W1a[i];
  }
  for (long i = tid; i < HID * HID; i += stride) {
    int k = (int)(i >> 7), n = (int)(i & 127);
    W1bT[n * HID + k] = (__bf16)W1b[i];
  }
  for (long i = tid; i < K2 * HID; i += stride) {
    int k = (int)(i >> 7), n = (int)(i & 127);
    W2aT[n * K2 + k] = (__bf16)W2a[i];
  }
  for (long i = tid; i < HID * HID; i += stride) {
    int k = (int)(i >> 7), n = (int)(i & 127);
    W2bT[n * HID + k] = (__bf16)W2b[i];
  }
}

// ---------------- Kernel 2: edge MLP + scatter-add ----------------
// h = relu([x[col]||e] @ W1a + b1a) @ W1b + b1b ; sums[row] += h ; cnt[row] += 1

__launch_bounds__(256, 1)
__global__ void edge_mlp_scatter(const __bf16* __restrict__ xb,
                                 const int*    __restrict__ eidx,   // [2*E]: row then col
                                 const float*  __restrict__ ea,     // [E, FE]
                                 const __bf16* __restrict__ W1aT, const float* __restrict__ b1a,
                                 const __bf16* __restrict__ W1bT, const float* __restrict__ b1b,
                                 float* __restrict__ sums, float* __restrict__ cnt,
                                 int N, int E, int nTiles)
{
  __shared__ __align__(16) __bf16 sW1a[HID * K1];    // transposed [n][k]
  __shared__ __align__(16) __bf16 sW1b[HID * HID];   // transposed [n][k]
  __shared__ float sB1a[HID];
  __shared__ float sB1b[HID];
  __shared__ __align__(16) __bf16 sA[8][16 * K1];    // per-wave A staging / h1 buffer

  const int tid  = threadIdx.x;
  const int wave = tid >> 5;
  const int lane = tid & 31;

  // Async bulk-copy pre-transposed bf16 weights into LDS (ASYNCcnt-tracked).
  {
    unsigned lw1a = lds_addr_of(sW1a);
    const char* g1a = (const char*)W1aT;
    for (int c = tid; c < (K1 * HID * 2) / 16; c += 256)
      async_copy16(lw1a + c * 16, g1a + c * 16);
    unsigned lw1b = lds_addr_of(sW1b);
    const char* g1b = (const char*)W1bT;
    for (int c = tid; c < (HID * HID * 2) / 16; c += 256)
      async_copy16(lw1b + c * 16, g1b + c * 16);
  }
  if (tid < HID) { sB1a[tid] = b1a[tid]; sB1b[tid] = b1b[tid]; }
  async_wait0();
  __syncthreads();

  __bf16* As = sA[wave];
  const unsigned asBase = lds_addr_of(As);
  const int nbase = lane & 15;
  const int mhalf = (lane >> 4) * 8;

  for (long tb = blockIdx.x; tb < nTiles; tb += gridDim.x) {
    long e0  = tb * 128 + (long)wave * 16;          // first edge of this wave's tile
    long e0c = e0;
    if (e0c > (long)E - 16) e0c = (long)E - 16;
    if (e0c < 0) e0c = 0;

    // ---- Stage A = [xb[col] || cvt(edge_attr)] as bf16, 16 x 192, two lanes per row ----
    {
      int m = lane >> 1, p = lane & 1;
      long eR = e0c + m; if (eR > (long)E - 1) eR = (long)E - 1;
      int  c  = eidx[(long)E + eR];                 // source node
      // x row: 128 bf16 = 256 B -> 16 chunks of 16 B, async global->LDS copy
      const char* xrow = (const char*)(xb + (long)c * FX);
      unsigned ldsRow  = asBase + m * (K1 * 2);
      #pragma unroll
      for (int q = 0; q < 8; ++q) {
        int ch = p * 8 + q;
        async_copy16(ldsRow + ch * 16, xrow + ch * 16);
      }
      // edge_attr: f32 stream, convert to bf16 inline (cols 128..191)
      const float* ep = ea + eR * (long)FE;
      __bf16* dst = As + m * K1 + FX;
      #pragma unroll
      for (int q = 0; q < 8; ++q) {
        int c4 = p * 8 + q;
        *(v4bf*)(dst + c4 * 4) = cvt4(*(const v4f*)(ep + c4 * 4));
      }
    }
    async_wait0();
    __syncthreads();

    // ---- Layer 1: [16x192] @ [192x128] ----
    v8f acc[8];
    #pragma unroll
    for (int t = 0; t < 8; ++t) acc[t] = (v8f){0,0,0,0,0,0,0,0};
    #pragma unroll
    for (int kb = 0; kb < K1; kb += 32) {
      v16bf a = ldsA_frag(As, K1, kb, lane);
      v16bf bf[8];
      #pragma unroll
      for (int t = 0; t < 8; ++t) bf[t] = ldsB_frag(sW1a, K1, kb, t, lane);
      #pragma unroll
      for (int t = 0; t < 8; ++t) acc[t] = wmma_bf16(a, bf[t], acc[t]);
    }
    // bias + relu, write h1 back into As as bf16 (16 x 128, stride HID)
    #pragma unroll
    for (int t = 0; t < 8; ++t) {
      float bias = sB1a[nbase + t * 16];
      #pragma unroll
      for (int r = 0; r < 8; ++r) {
        float v = acc[t][r] + bias;
        v = v > 0.0f ? v : 0.0f;
        As[(mhalf + r) * HID + nbase + t * 16] = (__bf16)v;
      }
    }
    __syncthreads();

    // ---- Layer 2: [16x128] @ [128x128] ----
    v8f acc2[8];
    #pragma unroll
    for (int t = 0; t < 8; ++t) acc2[t] = (v8f){0,0,0,0,0,0,0,0};
    #pragma unroll
    for (int kb = 0; kb < HID; kb += 32) {
      v16bf a = ldsA_frag(As, HID, kb, lane);
      v16bf bf[8];
      #pragma unroll
      for (int t = 0; t < 8; ++t) bf[t] = ldsB_frag(sW1b, HID, kb, t, lane);
      #pragma unroll
      for (int t = 0; t < 8; ++t) acc2[t] = wmma_bf16(a, bf[t], acc2[t]);
    }

    // ---- Scatter-add (f32 atomics into L2-resident sums) ----
    int rowIdx[8];
    #pragma unroll
    for (int r = 0; r < 8; ++r) {
      long eR = e0c + mhalf + r; if (eR > (long)E - 1) eR = (long)E - 1;
      rowIdx[r] = eidx[eR];                          // destination node
    }
    #pragma unroll
    for (int t = 0; t < 8; ++t) {
      float bias = sB1b[nbase + t * 16];
      #pragma unroll
      for (int r = 0; r < 8; ++r) {
        long eR = e0c + mhalf + r;
        if (eR >= e0 && eR < E) {
          atomicAdd(&sums[(long)rowIdx[r] * HID + nbase + t * 16], acc2[t][r] + bias);
        }
      }
    }
    if (lane < 16) {
      long eR = e0c + lane;
      if (eR >= e0 && eR < E) atomicAdd(&cnt[eidx[eR]], 1.0f);
    }
    __syncthreads();   // protect As before next iteration's restage
  }
}

// ---------------- Kernel 3: node MLP ----------------
// out = relu([x || sums/max(cnt,1)] @ W2a + b2a) @ W2b + b2b

__launch_bounds__(128, 1)
__global__ void node_mlp(const __bf16* __restrict__ xb,
                         const float* __restrict__ sums, const float* __restrict__ cnt,
                         const __bf16* __restrict__ W2aT, const float* __restrict__ b2a,
                         const __bf16* __restrict__ W2bT, const float* __restrict__ b2b,
                         float* __restrict__ out, int N, int nTiles)
{
  __shared__ __align__(16) __bf16 sW2a[HID * K2];    // transposed [n][k]
  __shared__ __align__(16) __bf16 sW2b[HID * HID];   // transposed [n][k]
  __shared__ float sB2a[HID];
  __shared__ float sB2b[HID];
  __shared__ __align__(16) __bf16 sA[4][16 * K2];

  const int tid  = threadIdx.x;
  const int wave = tid >> 5;
  const int lane = tid & 31;

  {
    unsigned lw2a = lds_addr_of(sW2a);
    const char* g2a = (const char*)W2aT;
    for (int c = tid; c < (K2 * HID * 2) / 16; c += 128)
      async_copy16(lw2a + c * 16, g2a + c * 16);
    unsigned lw2b = lds_addr_of(sW2b);
    const char* g2b = (const char*)W2bT;
    for (int c = tid; c < (HID * HID * 2) / 16; c += 128)
      async_copy16(lw2b + c * 16, g2b + c * 16);
  }
  if (tid < HID) { sB2a[tid] = b2a[tid]; sB2b[tid] = b2b[tid]; }
  async_wait0();
  __syncthreads();

  __bf16* As = sA[wave];
  const unsigned asBase = lds_addr_of(As);
  const int nbase = lane & 15;
  const int mhalf = (lane >> 4) * 8;

  for (long tb = blockIdx.x; tb < nTiles; tb += gridDim.x) {
    long i0  = tb * 64 + (long)wave * 16;
    long i0c = i0;
    if (i0c > (long)N - 16) i0c = (long)N - 16;
    if (i0c < 0) i0c = 0;

    // ---- Stage A = [xb || agg] as bf16, 16 x 256, two lanes per row ----
    {
      int m = lane >> 1, p = lane & 1;
      long iR = i0c + m; if (iR > (long)N - 1) iR = (long)N - 1;
      // x part: async 16B copies of the bf16 row
      const char* xrow = (const char*)(xb + iR * (long)FX);
      unsigned ldsRow  = asBase + m * (K2 * 2);
      #pragma unroll
      for (int q = 0; q < 8; ++q) {
        int ch = p * 8 + q;
        async_copy16(ldsRow + ch * 16, xrow + ch * 16);
      }
      // agg part: scaled f32 -> bf16
      const float* sp = sums + iR * (long)HID;
      float inv = 1.0f / fmaxf(cnt[iR], 1.0f);
      __bf16* dst = As + m * K2 + FX;
      #pragma unroll
      for (int q = 0; q < 16; ++q) {
        int c4 = p * 16 + q;
        v4f v = *(const v4f*)(sp + c4 * 4);
        v4f w = { v.x * inv, v.y * inv, v.z * inv, v.w * inv };
        *(v4bf*)(dst + c4 * 4) = cvt4(w);
      }
    }
    async_wait0();
    __syncthreads();

    // ---- Layer 1: [16x256] @ [256x128] ----
    v8f acc[8];
    #pragma unroll
    for (int t = 0; t < 8; ++t) acc[t] = (v8f){0,0,0,0,0,0,0,0};
    #pragma unroll
    for (int kb = 0; kb < K2; kb += 32) {
      v16bf a = ldsA_frag(As, K2, kb, lane);
      v16bf bf[8];
      #pragma unroll
      for (int t = 0; t < 8; ++t) bf[t] = ldsB_frag(sW2a, K2, kb, t, lane);
      #pragma unroll
      for (int t = 0; t < 8; ++t) acc[t] = wmma_bf16(a, bf[t], acc[t]);
    }
    #pragma unroll
    for (int t = 0; t < 8; ++t) {
      float bias = sB2a[nbase + t * 16];
      #pragma unroll
      for (int r = 0; r < 8; ++r) {
        float v = acc[t][r] + bias;
        v = v > 0.0f ? v : 0.0f;
        As[(mhalf + r) * HID + nbase + t * 16] = (__bf16)v;
      }
    }
    __syncthreads();

    // ---- Layer 2: [16x128] @ [128x128] ----
    v8f acc2[8];
    #pragma unroll
    for (int t = 0; t < 8; ++t) acc2[t] = (v8f){0,0,0,0,0,0,0,0};
    #pragma unroll
    for (int kb = 0; kb < HID; kb += 32) {
      v16bf a = ldsA_frag(As, HID, kb, lane);
      v16bf bf[8];
      #pragma unroll
      for (int t = 0; t < 8; ++t) bf[t] = ldsB_frag(sW2b, HID, kb, t, lane);
      #pragma unroll
      for (int t = 0; t < 8; ++t) acc2[t] = wmma_bf16(a, bf[t], acc2[t]);
    }

    // ---- Bias + store ----
    #pragma unroll
    for (int t = 0; t < 8; ++t) {
      float bias = sB2b[nbase + t * 16];
      #pragma unroll
      for (int r = 0; r < 8; ++r) {
        long iR = i0c + mhalf + r;
        if (iR >= i0 && iR < N) {
          out[iR * (long)FX + nbase + t * 16] = acc2[t][r] + bias;
        }
      }
    }
    __syncthreads();
  }
}

// ---------------- Host launcher ----------------

extern "C" void kernel_launch(void* const* d_in, const int* in_sizes, int n_in,
                              void* d_out, int out_size, void* d_ws, size_t ws_size,
                              hipStream_t stream) {
  const float* x    = (const float*)d_in[0];
  const int*   eidx = (const int*)  d_in[1];
  const float* ea   = (const float*)d_in[2];
  // d_in[3] = u, d_in[4] = batch : unused by the node model
  const float* W1a = (const float*)d_in[5];
  const float* b1a = (const float*)d_in[6];
  const float* W1b = (const float*)d_in[7];
  const float* b1b = (const float*)d_in[8];
  const float* W2a = (const float*)d_in[9];
  const float* b2a = (const float*)d_in[10];
  const float* W2b = (const float*)d_in[11];
  const float* b2b = (const float*)d_in[12];

  const int N = in_sizes[0] / FX;
  const int E = in_sizes[1] / 2;

  // Workspace layout (all 16B-aligned given 128-multiple N):
  char* w = (char*)d_ws;
  float*  sums = (float*)w;                                   // [N, HID] f32
  float*  cntp = (float*)(w + (long)N * HID * 4);             // [N] f32
  __bf16* xb   = (__bf16*)(w + (long)N * HID * 4 + (long)N * 4); // [N, FX] bf16
  __bf16* W1aT = xb + (long)N * FX;                           // [HID][K1]
  __bf16* W1bT = W1aT + K1 * HID;                             // [HID][HID]
  __bf16* W2aT = W1bT + HID * HID;                            // [HID][K2]
  __bf16* W2bT = W2aT + K2 * HID;                             // [HID][HID]

  prep_kernel<<<2048, 256, 0, stream>>>(x, W1a, W1b, W2a, W2b,
                                        xb, W1aT, W1bT, W2aT, W2bT,
                                        sums, cntp, N);

  // Edge phase: 128 edges per block-tile.
  int eTiles = (E + 127) / 128;
  int eGrid  = eTiles < 2048 ? eTiles : 2048;
  edge_mlp_scatter<<<eGrid, 256, 0, stream>>>(xb, eidx, ea, W1aT, b1a, W1bT, b1b,
                                              sums, cntp, N, E, eTiles);

  // Node phase: 64 nodes per block-tile.
  int nTiles = (N + 63) / 64;
  int nGrid  = nTiles < 1024 ? nTiles : 1024;
  node_mlp<<<nGrid, 128, 0, stream>>>(xb, sums, cntp, W2aT, b2a, W2bT, b2b,
                                      (float*)d_out, N, nTiles);
}
